// CNN_FINAL_53188874994263
// MI455X (gfx1250) — matrix operand order repaired and checked by reading, not compiled, not run on hardware
//
#include <hip/hip_runtime.h>
#include <hip/hip_bf16.h>

typedef __attribute__((ext_vector_type(16))) _Float16 v16h;
typedef __attribute__((ext_vector_type(8)))  float    v8f;

#define B_ 64
#define L_ 4096
#define V_ 200
#define H_ 256
#define NFILT 64
#define TOTALX ((long)B_ * L_ * V_)

// Per-gram (a = 2,3,4,5) constants: Kpad = roundup(a*200, 32)
// steps = Kpad/32 : {13, 19, 25, 32}; packed-weight offsets in halves.
#define NSTEPS_MAX 32

__device__ __forceinline__ void atomicMaxF(float* addr, float val) {
  // Sign-aware float max via integer atomics (works with -inf init).
  if (val >= 0.0f) atomicMax((int*)addr, __float_as_int(val));
  else             atomicMin((unsigned int*)addr, __float_as_uint(val));
}

__device__ __forceinline__ float sigm(float x) { return 1.0f / (1.0f + __expf(-x)); }

// ---------------------------------------------------------------------------
// One-time weight pre-pack: f32 (64,K) -> f16, zero-padded to Kpad, swizzled
// into per-lane WMMA B-fragment order:
//   pw[kb*2048 + nt*512 + lane*16 + h] = W[nt*16 + lane%16][kb*32 + (lane/16)*16 + h]
// ---------------------------------------------------------------------------
__global__ void pack_wgt(const float* __restrict__ wgt, _Float16* __restrict__ pw,
                         int K, int Kpad) {
  int t = blockIdx.x * blockDim.x + threadIdx.x;    // one thread per 8 halves
  if (t >= Kpad * 8) return;
  int flat = t * 8;
  int kb   = flat >> 11;
  int rem  = flat & 2047;
  int nt   = rem >> 9;
  int lane = (rem >> 4) & 31;
  int h0   = flat & 15;                             // 0 or 8
  int n    = nt * 16 + (lane & 15);
  int k0   = kb * 32 + ((lane >> 4) << 4) + h0;
  __align__(16) _Float16 hv[8];
  #pragma unroll
  for (int h = 0; h < 8; ++h) {
    int k = k0 + h;
    hv[h] = (k < K) ? (_Float16)wgt[(long)n * K + k] : (_Float16)0.0f;
  }
  *((uint4*)&pw[flat]) = *((uint4*)hv);
}

// ---------------------------------------------------------------------------
// Fused 4-gram conv (full-width V == GEMM) + max-pool. FP16 WMMA, f32 accum.
// One 256-thread block per (128-position tile, batch). The A matrix (window
// rows of x) is shared by all grams: x is streamed from HBM exactly ONCE.
// 16 accumulators: 4 grams x 4 filter-tiles. Grams drop out of the K-loop at
// their own Kpad (uniform scalar branch). Invalid tail rows masked at epilogue.
// ---------------------------------------------------------------------------
__device__ __forceinline__ void load_a_regs(const float* __restrict__ x, long aoff,
                                            int kkoff, float4 q[4]) {
  long gi = aoff + kkoff;
  if (gi > TOTALX - 16) gi = TOTALX - 16;           // never fault at buffer end
  const float4* xp = (const float4*)(x + gi);
  q[0] = xp[0]; q[1] = xp[1]; q[2] = xp[2]; q[3] = xp[3];
}

__device__ __forceinline__ void store_a_lds(_Float16* __restrict__ Ab, int ar, int akl,
                                            const float4 q[4]) {
  __align__(16) _Float16 hv[16];
  hv[0]=(_Float16)q[0].x; hv[1]=(_Float16)q[0].y; hv[2]=(_Float16)q[0].z; hv[3]=(_Float16)q[0].w;
  hv[4]=(_Float16)q[1].x; hv[5]=(_Float16)q[1].y; hv[6]=(_Float16)q[1].z; hv[7]=(_Float16)q[1].w;
  hv[8]=(_Float16)q[2].x; hv[9]=(_Float16)q[2].y; hv[10]=(_Float16)q[2].z; hv[11]=(_Float16)q[2].w;
  hv[12]=(_Float16)q[3].x; hv[13]=(_Float16)q[3].y; hv[14]=(_Float16)q[3].z; hv[15]=(_Float16)q[3].w;
  *((uint4*)&Ab[ar * 32 + akl])     = *((uint4*)hv);
  *((uint4*)&Ab[ar * 32 + akl + 8]) = *(((uint4*)hv) + 1);
}

__launch_bounds__(256)
__global__ void conv_pool_fused(const float* __restrict__ x,
                                const _Float16* __restrict__ pw,  // packed B (all grams)
                                float* __restrict__ pool)         // (B,256): g*64+f
{
  __shared__ __align__(32) _Float16 As[2][128 * 32];   // 2 x 8KB A tiles

  const int tid  = threadIdx.x;
  const int b    = blockIdx.y;
  const int tile = blockIdx.x;
  const long xbase = (long)b * (L_ * (long)V_);

  // A-tile load mapping: 2 threads/row, 16 consecutive k (f32->f16)
  const int ar  = tid >> 1;
  const int akl = (tid & 1) << 4;
  const long aoff = xbase + (long)(tile * 128 + ar) * V_ + akl;

  // WMMA fragment mapping (contiguous k = 16*(lane/16) + s, same order as pack_wgt)
  const int lane = tid & 31;
  const int wv   = tid >> 5;
  const int frow = wv * 16 + (lane & 15);
  const int fk   = (lane >> 4) << 4;                   // 0 or 16

  const int  STEPS[4] = {13, 19, 25, 32};              // Kpad/32 per gram
  const long PWOFF[4] = {0, 26624, 65536, 116736};     // halves

  v8f acc[4][4] = {};

  // Prologue: stage step 0
  {
    float4 q[4];
    load_a_regs(x, aoff, 0, q);
    store_a_lds(As[0], ar, akl, q);
  }

  for (int s = 0; s < NSTEPS_MAX; ++s) {
    __syncthreads();                                   // As[s&1] visible to all
    const bool have_next = (s + 1) < NSTEPS_MAX;
    float4 q[4];
    if (have_next) load_a_regs(x, aoff, (s + 1) << 5, q);   // overlap with WMMA

    const _Float16* Ab = As[s & 1];
    v16h av;
    *((uint4*)&av)       = *((const uint4*)&Ab[frow * 32 + fk]);
    *(((uint4*)&av) + 1) = *((const uint4*)&Ab[frow * 32 + fk + 8]);

    #pragma unroll
    for (int g = 0; g < 4; ++g) {
      if (s < STEPS[g]) {                              // uniform scalar branch
        const _Float16* Bp = pw + PWOFF[g] + ((long)s << 11);
        #pragma unroll
        for (int nt = 0; nt < 4; ++nt) {
          v16h bv = *((const v16h*)&Bp[nt * 512 + lane * 16]);
          acc[g][nt] = __builtin_amdgcn_wmma_f32_16x16x32_f16(
              false, av, false, bv, (short)0, acc[g][nt], false, false);
        }
      }
    }

    if (have_next) store_a_lds(As[(s + 1) & 1], ar, akl, q);
  }

  // ---- masked max over positions, all 4 grams, then one atomic per column ----
  const int rowbase = tile * 128 + wv * 16 + ((lane >> 4) << 3);
  __syncthreads();
  float* Red = (float*)As;                 // reuse LDS: 8 waves x 256 cols
  #pragma unroll
  for (int g = 0; g < 4; ++g) {
    const int Pg = 4095 - g;               // P = L - gram + 1, gram = g+2
    #pragma unroll
    for (int nt = 0; nt < 4; ++nt) {
      float v = __int_as_float(0xff800000);
      #pragma unroll
      for (int r = 0; r < 8; ++r)
        if (rowbase + r < Pg) v = fmaxf(v, acc[g][nt][r]);
      v = fmaxf(v, __shfl_xor(v, 16, 32)); // combine row-halves (wave32)
      if (lane < 16) Red[wv * 256 + g * 64 + nt * 16 + lane] = v;
    }
  }
  __syncthreads();
  {
    float v = Red[tid];
    #pragma unroll
    for (int w2 = 1; w2 < 8; ++w2) v = fmaxf(v, Red[w2 * 256 + tid]);
    atomicMaxF(&pool[(long)b * 256 + tid], v);
  }
}

// ---------------------------------------------------------------------------
// Small f32 kernels for the LSTM/FC tail (negligible FLOPs vs conv stage)
// ---------------------------------------------------------------------------
__global__ void init_ws(float* pool, float* zbuf) {
  int i = blockIdx.x * blockDim.x + threadIdx.x;
  if (i < B_ * H_) {
    pool[i] = __int_as_float(0xff800000);  // -inf
    #pragma unroll
    for (int j = 0; j < 6; ++j) zbuf[(long)j * B_ * H_ + i] = 0.0f;
  }
}

__global__ void fc_add_bias(float* fc, const float* b0, const float* b1,
                            const float* b2, const float* b3) {
  int i = blockIdx.x * blockDim.x + threadIdx.x;
  if (i >= B_ * H_) return;
  int r = i & 255, gi = r >> 6, f = r & 63;
  const float* bp = (gi == 0) ? b0 : (gi == 1) ? b1 : (gi == 2) ? b2 : b3;
  fc[i] += bp[f];
}

__launch_bounds__(256)
__global__ void lstm_cell_k(const float* __restrict__ fc,
                            const float* __restrict__ hprev, const float* __restrict__ cprev,
                            const float* __restrict__ Wih, const float* __restrict__ Whh,
                            const float* __restrict__ bih, const float* __restrict__ bhh,
                            float* __restrict__ hout, float* __restrict__ cout) {
  const int b = blockIdx.x, j = threadIdx.x;
  __shared__ float xs[H_], hs[H_];
  xs[j] = fc[b * H_ + j];
  hs[j] = hprev[b * H_ + j];
  __syncthreads();
  float gte[4];
  #pragma unroll
  for (int gi2 = 0; gi2 < 4; ++gi2) {
    int r = gi2 * H_ + j;
    float acc = bih[r] + bhh[r];
    const float* wi = Wih + (long)r * H_;
    const float* wh = Whh + (long)r * H_;
    for (int k = 0; k < H_; ++k) acc = fmaf(xs[k], wi[k], fmaf(hs[k], wh[k], acc));
    gte[gi2] = acc;
  }
  float cn = sigm(gte[1]) * cprev[b * H_ + j] + sigm(gte[0]) * tanhf(gte[2]);
  float hn = sigm(gte[3]) * tanhf(cn);
  hout[b * H_ + j] = hn;
  cout[b * H_ + j] = cn;
}

__global__ void copy_hc(const float* h, const float* c,
                        float* h2, float* c2, float* h3, float* c3) {
  int i = blockIdx.x * blockDim.x + threadIdx.x;
  if (i < B_ * H_) { float hv = h[i], cv = c[i]; h2[i] = hv; h3[i] = hv; c2[i] = cv; c3[i] = cv; }
}

__launch_bounds__(256)
__global__ void accum_hc(const float* __restrict__ hin, const float* __restrict__ cin,
                         const float* __restrict__ hw, const float* __restrict__ hb,
                         const float* __restrict__ cw, const float* __restrict__ cb,
                         float* __restrict__ hout, float* __restrict__ cout) {
  const int b = blockIdx.x, j = threadIdx.x;
  __shared__ float hs[H_], cs[H_];
  hs[j] = hin[b * H_ + j];
  cs[j] = cin[b * H_ + j];
  __syncthreads();
  float ha = hb[j], ca = cb[j];
  const float* hwr = hw + (long)j * H_;
  const float* cwr = cw + (long)j * H_;
  for (int k = 0; k < H_; ++k) { ha = fmaf(hs[k], hwr[k], ha); ca = fmaf(cs[k], cwr[k], ca); }
  hout[b * H_ + j] += ha;
  cout[b * H_ + j] += ca;
}

__launch_bounds__(256)
__global__ void head_k(const float* __restrict__ h,
                       const float* __restrict__ mw, const float* __restrict__ mb,
                       const float* __restrict__ ow, const float* __restrict__ ob,
                       float* __restrict__ out, int C) {
  const int b = blockIdx.x, j = threadIdx.x;
  __shared__ float hs[H_], ms[H_];
  hs[j] = h[b * H_ + j];
  __syncthreads();
  float acc = mb[j];
  const float* mwr = mw + (long)j * H_;
  for (int k = 0; k < H_; ++k) acc = fmaf(hs[k], mwr[k], acc);
  ms[j] = fmaxf(acc, 0.0f);
  __syncthreads();
  if (j < C) {
    float a2 = ob[j];
    const float* owr = ow + (long)j * H_;
    for (int k = 0; k < H_; ++k) a2 = fmaf(ms[k], owr[k], a2);
    out[(long)b * C + j] = a2;
  }
}

// ---------------------------------------------------------------------------
extern "C" void kernel_launch(void* const* d_in, const int* in_sizes, int n_in,
                              void* d_out, int out_size, void* d_ws, size_t ws_size,
                              hipStream_t stream) {
  const float *x, *convw[4], *convb[4], *Wih, *Whh, *bih, *bhh;
  const float *hfcw, *hfcb, *cfcw, *cfcb, *midw, *midb, *outw[3], *outb[3];

  if (n_in >= 26) {  // python lists expanded into individual inputs
    x = (const float*)d_in[0];
    for (int i = 0; i < 4; ++i) { convw[i] = (const float*)d_in[2 + i]; convb[i] = (const float*)d_in[6 + i]; }
    Wih  = (const float*)d_in[10]; Whh  = (const float*)d_in[11];
    bih  = (const float*)d_in[12]; bhh  = (const float*)d_in[13];
    hfcw = (const float*)d_in[14]; hfcb = (const float*)d_in[15];
    cfcw = (const float*)d_in[16]; cfcb = (const float*)d_in[17];
    midw = (const float*)d_in[18]; midb = (const float*)d_in[19];
    for (int i = 0; i < 3; ++i) { outw[i] = (const float*)d_in[20 + i]; outb[i] = (const float*)d_in[23 + i]; }
  } else {           // lists passed as single concatenated buffers (16 inputs)
    x = (const float*)d_in[0];
    const float* cw = (const float*)d_in[2];
    const float* cb = (const float*)d_in[3];
    long o = 0;
    for (int i = 0; i < 4; ++i) { convw[i] = cw + o; o += (long)NFILT * (2 + i) * V_; convb[i] = cb + i * NFILT; }
    Wih  = (const float*)d_in[4];  Whh  = (const float*)d_in[5];
    bih  = (const float*)d_in[6];  bhh  = (const float*)d_in[7];
    hfcw = (const float*)d_in[8];  hfcb = (const float*)d_in[9];
    cfcw = (const float*)d_in[10]; cfcb = (const float*)d_in[11];
    midw = (const float*)d_in[12]; midb = (const float*)d_in[13];
    const float* ow  = (const float*)d_in[14];
    const float* obp = (const float*)d_in[15];
    const int Cc[3] = {183, 202, 11};
    long wo = 0, bo = 0;
    for (int i = 0; i < 3; ++i) { outw[i] = ow + wo; outb[i] = obp + bo; wo += (long)Cc[i] * H_; bo += Cc[i]; }
  }

  float* ws = (float*)d_ws;
  float* fc  = ws;                       // (B, 256) pooled feats -> fc_input
  float* hid = ws + B_ * H_;             // 6 hidden buffers h1,c1,h2,c2,h3,c3
  float* hid_h[4] = {nullptr, hid + 0 * B_ * H_, hid + 2 * B_ * H_, hid + 4 * B_ * H_};
  float* hid_c[4] = {nullptr, hid + 1 * B_ * H_, hid + 3 * B_ * H_, hid + 5 * B_ * H_};
  float* out_h = hid + 6 * B_ * H_;
  float* out_c = hid + 7 * B_ * H_;
  _Float16* pwbase = (_Float16*)(ws + 9 * B_ * H_);   // packed weights region
  float* outp  = (float*)d_out;

  init_ws<<<dim3((B_ * H_ + 255) / 256), dim3(256), 0, stream>>>(fc, hid);

  // Pack all 4 grams' weights (offsets must match PWOFF in conv_pool_fused)
  const int grams[4] = {2, 3, 4, 5};
  const long pwoff[4] = {0, 26624, 65536, 116736};
  for (int g = 0; g < 4; ++g) {
    int K = grams[g] * V_;
    int Kpad = (K + 31) & ~31;
    pack_wgt<<<dim3((Kpad * 8 + 255) / 256), dim3(256), 0, stream>>>(
        convw[g], pwbase + pwoff[g], K, Kpad);
  }
  // Fused conv+pool: x streamed once for all 4 grams
  conv_pool_fused<<<dim3(32, B_), dim3(256), 0, stream>>>(x, pwbase, fc);

  fc_add_bias<<<dim3(64), dim3(256), 0, stream>>>(fc, convb[0], convb[1], convb[2], convb[3]);

  const int Cs[3] = {183, 202, 11};
  long ooff = 0;
  for (int a = 1; a <= 3; ++a) {
    lstm_cell_k<<<dim3(B_), dim3(H_), 0, stream>>>(
        fc, hid_h[a], hid_c[a],
        Wih + (long)(a - 1) * 4 * H_ * H_, Whh + (long)(a - 1) * 4 * H_ * H_,
        bih + (long)(a - 1) * 4 * H_,     bhh + (long)(a - 1) * 4 * H_,
        out_h, out_c);
    if (a == 1) {
      copy_hc<<<dim3(64), dim3(256), 0, stream>>>(out_h, out_c,
                                                  hid_h[2], hid_c[2], hid_h[3], hid_c[3]);
    } else if (a == 2) {
      const long st = (long)(2 * 4 + 3);  // [a=2][t=3] slice of (4,4,...) arrays
      accum_hc<<<dim3(B_), dim3(H_), 0, stream>>>(
          out_h, out_c,
          hfcw + st * H_ * H_, hfcb + st * H_,
          cfcw + st * H_ * H_, cfcb + st * H_,
          hid_h[3], hid_c[3]);
    }
    head_k<<<dim3(B_), dim3(H_), 0, stream>>>(
        out_h, midw + (long)(a - 1) * H_ * H_, midb + (long)(a - 1) * H_,
        outw[a - 1], outb[a - 1], outp + ooff, Cs[a - 1]);
    ooff += (long)B_ * Cs[a - 1];
  }
}
